// GraphTransformerDetector_91276644974954
// MI455X (gfx1250) — matrix operand order repaired and checked by reading, not compiled, not run on hardware
//
#include <hip/hip_runtime.h>
#include <hip/hip_bf16.h>
#include <math.h>

typedef __attribute__((ext_vector_type(4)))  __bf16 v4bf;
typedef __attribute__((ext_vector_type(8)))  __bf16 v8bf;
typedef __attribute__((ext_vector_type(16))) __bf16 v16bf;
typedef __attribute__((ext_vector_type(8)))  float  v8f;

#define WAVES_PER_BLOCK 8
#define BM 128
#define BN 32
#define BK 32
#define LDA 40   // padded LDS stride (bf16 elems) -> bank spread, 16B aligned rows
#define LDB 40

// ---------------------------------------------------------------------------
// Tiled WMMA GEMM: C[M,Nout] = act(A[M,K] @ W[K,Nld] + bias)
// Requirements (guaranteed by caller via padded buffers):
//   K % 32 == 0 (A leading dim == K), Nld % 32 == 0 (W leading dim == Nld)
// Nout <= Nld is the real output width / C stride (write-guarded).
// 256 threads = 8 waves; block tile 128x32; each wave: one 16-row A fragment
// reused across two v_wmma_f32_16x16x32_bf16 (two 16-col tiles).
// ACT: 0=none, 1=relu, 2=gelu(exact)
// ---------------------------------------------------------------------------
template <int ACT>
__global__ __launch_bounds__(256) void gemm_wmma_bf16(
    const float* __restrict__ A, const float* __restrict__ W,
    const float* __restrict__ bias, float* __restrict__ C,
    int M, int K, int Nld, int Nout)
{
    __shared__ __bf16 As[BM * LDA];   // ~10 KB
    __shared__ __bf16 Bs[BN * LDB];   // ~2.5 KB, stored transposed: Bs[n*LDB+k]

    const int tid  = threadIdx.x;
    const int wave = tid >> 5;
    const int lane = tid & 31;
    const int bm   = blockIdx.x * BM;
    const int bn   = blockIdx.y * BN;
    const int m    = lane & 15;
    const int half = lane >> 4;

    v8f acc0 = {}, acc1 = {};

    for (int k0 = 0; k0 < K; k0 += BK) {
        // ---- A tile: 128x32 f32, 4 x float4 per thread, row-clamped (no branches)
        #pragma unroll
        for (int j = 0; j < 4; ++j) {
            int q  = tid + j * 256;
            int r  = q >> 3;
            int cv = (q & 7) << 2;
            int gr = bm + r; gr = gr < M ? gr : M - 1;
            const float4 f = *(const float4*)&A[(size_t)gr * K + k0 + cv];
            v4bf b;
            b[0] = (__bf16)f.x; b[1] = (__bf16)f.y;
            b[2] = (__bf16)f.z; b[3] = (__bf16)f.w;
            *(v4bf*)&As[r * LDA + cv] = b;
        }
        // ---- W tile: 32x32 f32, transposed into LDS (Bs[n*LDB+k])
        #pragma unroll
        for (int j = 0; j < 2; ++j) {
            int p = (tid + j * 256) << 1;
            int r = p >> 5;          // k within tile
            int c = p & 31;          // n within tile (even)
            const float2 f = *(const float2*)&W[(size_t)(k0 + r) * Nld + bn + c];
            Bs[c * LDB + r]       = (__bf16)f.x;
            Bs[(c + 1) * LDB + r] = (__bf16)f.y;
        }
        __syncthreads();

        // prefetch next A tile into L2 (global_prefetch_b8)
        if (k0 + BK < K) {
            int pr = bm + (tid >> 1); pr = pr < M ? pr : M - 1;
            __builtin_prefetch(&A[(size_t)pr * K + k0 + BK + (tid & 1) * 16], 0, 1);
        }

        // ---- fragments (contiguous runs -> ds_load_b128 pairs)
        const int arow = wave * 16 + m;
        v8bf alo = *(const v8bf*)&As[arow * LDA + half * 16];
        v8bf ahi = *(const v8bf*)&As[arow * LDA + half * 16 + 8];
        v16bf aF = __builtin_shufflevector(alo, ahi,
                        0,1,2,3,4,5,6,7,8,9,10,11,12,13,14,15);

        v8bf b0lo = *(const v8bf*)&Bs[m * LDB + half * 16];
        v8bf b0hi = *(const v8bf*)&Bs[m * LDB + half * 16 + 8];
        v16bf bF0 = __builtin_shufflevector(b0lo, b0hi,
                        0,1,2,3,4,5,6,7,8,9,10,11,12,13,14,15);

        v8bf b1lo = *(const v8bf*)&Bs[(16 + m) * LDB + half * 16];
        v8bf b1hi = *(const v8bf*)&Bs[(16 + m) * LDB + half * 16 + 8];
        v16bf bF1 = __builtin_shufflevector(b1lo, b1hi,
                        0,1,2,3,4,5,6,7,8,9,10,11,12,13,14,15);

        acc0 = __builtin_amdgcn_wmma_f32_16x16x32_bf16(
                   false, aF, false, bF0, (short)0, acc0, false, false);
        acc1 = __builtin_amdgcn_wmma_f32_16x16x32_bf16(
                   false, aF, false, bF1, (short)0, acc1, false, false);
        __syncthreads();
    }

    // ---- epilogue: C layout rows = half*8 + r, col = lane&15
    const int col   = lane & 15;
    const int rbase = half * 8;
    #pragma unroll
    for (int r = 0; r < 8; ++r) {
        int grow = bm + wave * 16 + rbase + r;
        if (grow >= M) continue;
        #pragma unroll
        for (int t = 0; t < 2; ++t) {
            int gcol = bn + t * 16 + col;
            if (gcol < Nout) {
                float v = (t == 0) ? acc0[r] : acc1[r];
                if (bias) v += bias[gcol];
                if (ACT == 1) v = fmaxf(v, 0.0f);
                else if (ACT == 2)
                    v = 0.5f * v * (1.0f + erff(v * 0.70710678118654752f));
                C[(size_t)grow * Nout + gcol] = v;
            }
        }
    }
}

// ---------------------------------------------------------------------------
// LayerNorm over D=128, wave per row, float4 vectorized.
// Y = [relu]( LN(A [+ B]) * g + b )
// ---------------------------------------------------------------------------
__global__ __launch_bounds__(256) void ln128_kernel(
    const float* __restrict__ A, const float* __restrict__ B,
    const float* __restrict__ g, const float* __restrict__ bb,
    float* __restrict__ Y, int n, int relu)
{
    const int wave = threadIdx.x >> 5, lane = threadIdx.x & 31;
    const int row  = blockIdx.x * WAVES_PER_BLOCK + wave;
    if (row >= n) return;

    float4 v = *(const float4*)&A[(size_t)row * 128 + lane * 4];
    if (B) {
        float4 u = *(const float4*)&B[(size_t)row * 128 + lane * 4];
        v.x += u.x; v.y += u.y; v.z += u.z; v.w += u.w;
    }
    float s = v.x + v.y + v.z + v.w;
    #pragma unroll
    for (int off = 16; off; off >>= 1) s += __shfl_xor(s, off, 32);
    float mean = s * (1.0f / 128.0f);
    float qx = v.x - mean, qy = v.y - mean, qz = v.z - mean, qw = v.w - mean;
    float q = qx * qx + qy * qy + qz * qz + qw * qw;
    #pragma unroll
    for (int off = 16; off; off >>= 1) q += __shfl_xor(q, off, 32);
    float inv = rsqrtf(q * (1.0f / 128.0f) + 1e-5f);

    float4 gg = *(const float4*)&g[lane * 4];
    float4 bbv = *(const float4*)&bb[lane * 4];
    float4 o;
    o.x = qx * inv * gg.x + bbv.x;
    o.y = qy * inv * gg.y + bbv.y;
    o.z = qz * inv * gg.z + bbv.z;
    o.w = qw * inv * gg.w + bbv.w;
    if (relu) {
        o.x = fmaxf(o.x, 0.0f); o.y = fmaxf(o.y, 0.0f);
        o.z = fmaxf(o.z, 0.0f); o.w = fmaxf(o.w, 0.0f);
    }
    *(float4*)&Y[(size_t)row * 128 + lane * 4] = o;
}

// ---------------------------------------------------------------------------
// small elementwise / scatter kernels
// ---------------------------------------------------------------------------

// zero-pad weight [srcR,srcC] into [dstR,dstC]
__global__ void pad_weight_kernel(const float* __restrict__ src,
                                  float* __restrict__ dst,
                                  int srcR, int srcC, int dstR, int dstC)
{
    int i = blockIdx.x * blockDim.x + threadIdx.x;
    if (i >= dstR * dstC) return;
    int r = i / dstC, c = i % dstC;
    dst[i] = (r < srcR && c < srcC) ? src[r * srcC + c] : 0.0f;
}

// X0[n,32] = [xyz, attr, 0...]
__global__ void build_x0_kernel(const float* __restrict__ xyz,
                                const float* __restrict__ attr,
                                float* __restrict__ X0, int n)
{
    int i = blockIdx.x * blockDim.x + threadIdx.x;
    if (i >= n) return;
    float4 v = make_float4(xyz[i * 3], xyz[i * 3 + 1], xyz[i * 3 + 2], attr[i]);
    *(float4*)&X0[(size_t)i * 32] = v;
    float4 z = make_float4(0.f, 0.f, 0.f, 0.f);
    #pragma unroll
    for (int j = 1; j < 8; ++j) *(float4*)&X0[(size_t)i * 32 + j * 4] = z;
}

// rel[e,32] = [xyz[dst]-xyz[src], 0...]
__global__ void build_rel_kernel(const float* __restrict__ xyz,
                                 const int* __restrict__ ei,
                                 float* __restrict__ rel, int ne)
{
    int e = blockIdx.x * blockDim.x + threadIdx.x;
    if (e >= ne) return;
    int s = ei[e], d = ei[ne + e];
    float4 v = make_float4(xyz[d * 3 + 0] - xyz[s * 3 + 0],
                           xyz[d * 3 + 1] - xyz[s * 3 + 1],
                           xyz[d * 3 + 2] - xyz[s * 3 + 2], 0.f);
    *(float4*)&rel[(size_t)e * 32] = v;
    float4 z = make_float4(0.f, 0.f, 0.f, 0.f);
    #pragma unroll
    for (int j = 1; j < 8; ++j) *(float4*)&rel[(size_t)e * 32 + j * 4] = z;
}

__global__ void zero_kernel(float* __restrict__ p, size_t n)
{
    size_t i = (size_t)blockIdx.x * blockDim.x + threadIdx.x;
    if (i < n) p[i] = 0.0f;
}

__global__ void init_softmax_kernel(float* __restrict__ amax,
                                    float* __restrict__ den, int n4)
{
    int i = blockIdx.x * blockDim.x + threadIdx.x;
    if (i >= n4) return;
    amax[i] = -3.402823466e38f;
    den[i]  = 0.0f;
}

// alpha[e,h] = (q[dst,h,:] . (k[src,h,:] + ee[e,h,:])) / sqrt(32)
__global__ void attn_alpha_kernel(const float* __restrict__ q,
                                  const float* __restrict__ k,
                                  const float* __restrict__ ee,
                                  const int* __restrict__ ei,
                                  float* __restrict__ alpha, int ne)
{
    int idx = blockIdx.x * blockDim.x + threadIdx.x;
    if (idx >= ne * 4) return;
    int e = idx >> 2, h = idx & 3;
    int s = ei[e], d = ei[ne + e];
    const float* qp = q  + (size_t)d * 128 + h * 32;
    const float* kp = k  + (size_t)s * 128 + h * 32;
    const float* ep = ee + (size_t)e * 128 + h * 32;
    float acc = 0.0f;
    #pragma unroll
    for (int c = 0; c < 32; c += 4) {
        float4 a = *(const float4*)&qp[c];
        float4 b = *(const float4*)&kp[c];
        float4 g = *(const float4*)&ep[c];
        acc += a.x * (b.x + g.x) + a.y * (b.y + g.y)
             + a.z * (b.z + g.z) + a.w * (b.w + g.w);
    }
    alpha[idx] = acc * 0.17677669529663687f;  // 1/sqrt(32)
}

static __device__ __forceinline__ void atomicMaxFloat(float* addr, float val)
{
    int* ia  = (int*)addr;
    int  old = __float_as_int(*addr);
    while (__int_as_float(old) < val) {
        int assumed = old;
        old = atomicCAS(ia, assumed, __float_as_int(val));
        if (old == assumed) break;
    }
}

__global__ void seg_max_kernel(const float* __restrict__ alpha,
                               const int* __restrict__ ei,
                               float* __restrict__ amax, int ne)
{
    int idx = blockIdx.x * blockDim.x + threadIdx.x;
    if (idx >= ne * 4) return;
    int e = idx >> 2, h = idx & 3;
    int d = ei[ne + e];
    atomicMaxFloat(&amax[d * 4 + h], alpha[idx]);
}

// alpha := exp(alpha - amax[dst]); den[dst] += alpha
__global__ void seg_expsum_kernel(float* __restrict__ alpha,
                                  const int* __restrict__ ei,
                                  const float* __restrict__ amax,
                                  float* __restrict__ den, int ne)
{
    int idx = blockIdx.x * blockDim.x + threadIdx.x;
    if (idx >= ne * 4) return;
    int e = idx >> 2, h = idx & 3;
    int d = ei[ne + e];
    float ex = __expf(alpha[idx] - amax[d * 4 + h]);
    alpha[idx] = ex;
    atomicAdd(&den[d * 4 + h], ex);
}

// out[dst, c] += (v[src,c] + ee[e,c]) * alpha[e,h]/den[dst,h]
__global__ void attn_msg_kernel(const float* __restrict__ v,
                                const float* __restrict__ ee,
                                const float* __restrict__ alpha,
                                const float* __restrict__ den,
                                const int* __restrict__ ei,
                                float* __restrict__ out, int ne)
{
    size_t idx = (size_t)blockIdx.x * blockDim.x + threadIdx.x;
    if (idx >= (size_t)ne * 128) return;
    int e = (int)(idx >> 7), c = (int)(idx & 127), h = c >> 5;
    int s = ei[e], d = ei[ne + e];
    float a = alpha[e * 4 + h] / den[d * 4 + h];
    float m = (v[(size_t)s * 128 + c] + ee[(size_t)e * 128 + c]) * a;
    atomicAdd(&out[(size_t)d * 128 + c], m);
}

// beta = sigmoid([out, r, out-r] @ Wbeta); out := beta*r + (1-beta)*out
__global__ __launch_bounds__(256) void beta_gate_kernel(
    float* __restrict__ out, const float* __restrict__ r,
    const float* __restrict__ Wb, int n)
{
    const int wave = threadIdx.x >> 5, lane = threadIdx.x & 31;
    const int row  = blockIdx.x * WAVES_PER_BLOCK + wave;
    if (row >= n) return;
    float s = 0.0f;
    #pragma unroll
    for (int j = 0; j < 4; ++j) {
        int i = lane + j * 32;
        float o  = out[(size_t)row * 128 + i];
        float rr = r[(size_t)row * 128 + i];
        s += o * Wb[i] + rr * Wb[128 + i] + (o - rr) * Wb[256 + i];
    }
    #pragma unroll
    for (int off = 16; off; off >>= 1) s += __shfl_xor(s, off, 32);
    float beta = 1.0f / (1.0f + __expf(-s));
    #pragma unroll
    for (int j = 0; j < 4; ++j) {
        int i = lane + j * 32;
        float o  = out[(size_t)row * 128 + i];
        float rr = r[(size_t)row * 128 + i];
        out[(size_t)row * 128 + i] = beta * rr + (1.0f - beta) * o;
    }
}

// ---------------------------------------------------------------------------
// host orchestration
// ---------------------------------------------------------------------------
static inline int cdiv(int a, int b) { return (a + b - 1) / b; }

template <int ACT>
static void launch_gemm(const float* A, const float* W, const float* bias,
                        float* C, int M, int K, int Nld, int Nout, hipStream_t s)
{
    dim3 grid(cdiv(M, BM), Nld / BN);
    gemm_wmma_bf16<ACT><<<grid, 256, 0, s>>>(A, W, bias, C, M, K, Nld, Nout);
}

extern "C" void kernel_launch(void* const* d_in, const int* in_sizes, int n_in,
                              void* d_out, int out_size, void* d_ws, size_t ws_size,
                              hipStream_t stream)
{
    (void)n_in; (void)out_size; (void)ws_size;

    const float* xyz    = (const float*)d_in[0];
    const float* attr   = (const float*)d_in[1];
    const int*   ei     = (const int*)  d_in[2];
    const float* enc_W  = (const float*)d_in[3];
    const float* enc_b  = (const float*)d_in[4];
    const float* enc_g  = (const float*)d_in[5];
    const float* enc_bb = (const float*)d_in[6];
    const float* pos_W1 = (const float*)d_in[7];
    const float* pos_b1 = (const float*)d_in[8];
    const float* pos_g  = (const float*)d_in[9];
    const float* pos_bb = (const float*)d_in[10];
    const float* pos_W2 = (const float*)d_in[11];
    const float* pos_b2 = (const float*)d_in[12];
    const float* Wq     = (const float*)d_in[13];
    const float* bq     = (const float*)d_in[14];
    const float* Wk     = (const float*)d_in[15];
    const float* bk     = (const float*)d_in[16];
    const float* Wv     = (const float*)d_in[17];
    const float* bv     = (const float*)d_in[18];
    const float* We     = (const float*)d_in[19];
    const float* Wskip  = (const float*)d_in[20];
    const float* bskip  = (const float*)d_in[21];
    const float* Wbeta  = (const float*)d_in[22];
    const float* ln1_g  = (const float*)d_in[23];
    const float* ln1_b  = (const float*)d_in[24];
    const float* ln2_g  = (const float*)d_in[25];
    const float* ln2_b  = (const float*)d_in[26];
    const float* fW1    = (const float*)d_in[27];
    const float* fb1    = (const float*)d_in[28];
    const float* fW2    = (const float*)d_in[29];
    const float* fb2    = (const float*)d_in[30];
    const float* cls_W1 = (const float*)d_in[31];
    const float* cls_b1 = (const float*)d_in[32];
    const float* cls_W2 = (const float*)d_in[33];
    const float* cls_b2 = (const float*)d_in[34];
    const float* box_W1 = (const float*)d_in[35];
    const float* box_b1 = (const float*)d_in[36];
    const float* box_W2 = (const float*)d_in[37];
    const float* box_b2 = (const float*)d_in[38];

    const int N = in_sizes[0] / 3;
    const int E = in_sizes[2] / 2;

    // workspace carve (floats)
    float* ws = (float*)d_ws;
    float* X0     = ws; ws += (size_t)N * 32;   // padded encoder input
    float* x      = ws; ws += (size_t)N * 128;
    float* tmp    = ws; ws += (size_t)N * 256;  // FFN hidden / pre-LN scratch
    float* rel    = ws; ws += (size_t)E * 32;   // padded rel-pos input
    float* eA     = ws; ws += (size_t)E * 128;  // pos hidden, then per-layer ee
    float* eF     = ws; ws += (size_t)E * 128;  // persistent edge features
    float* qb     = ws; ws += (size_t)N * 128;
    float* kb     = ws; ws += (size_t)N * 128;
    float* vb     = ws; ws += (size_t)N * 128;
    float* rb     = ws; ws += (size_t)N * 128;
    float* agg    = ws; ws += (size_t)N * 128;
    float* alpha  = ws; ws += (size_t)E * 4;
    float* amax   = ws; ws += (size_t)N * 4;
    float* den    = ws; ws += (size_t)N * 4;
    float* h64    = ws; ws += (size_t)N * 64;
    float* encWp  = ws; ws += 32 * 128;         // padded weights
    float* posW1p = ws; ws += 32 * 128;
    float* clsW2p = ws; ws += 64 * 32;
    float* boxW2p = ws; ws += 64 * 32;

    float* cls_out = (float*)d_out;
    float* box_out = cls_out + (size_t)N * 4;

    const int TB = 256;

    // ---- pad irregular weights so every GEMM is guard-free in the hot loop
    pad_weight_kernel<<<cdiv(32 * 128, TB), TB, 0, stream>>>(enc_W,  encWp,  4, 128, 32, 128);
    pad_weight_kernel<<<cdiv(32 * 128, TB), TB, 0, stream>>>(pos_W1, posW1p, 3, 128, 32, 128);
    pad_weight_kernel<<<cdiv(64 * 32, TB), TB, 0, stream>>>(cls_W2, clsW2p, 64, 4, 64, 32);
    pad_weight_kernel<<<cdiv(64 * 32, TB), TB, 0, stream>>>(box_W2, boxW2p, 64, 7, 64, 32);

    // ---- input encoder: Linear(4->128) -> LN -> ReLU
    build_x0_kernel<<<cdiv(N, TB), TB, 0, stream>>>(xyz, attr, X0, N);
    launch_gemm<0>(X0, encWp, enc_b, tmp, N, 32, 128, 128, stream);
    ln128_kernel<<<cdiv(N, WAVES_PER_BLOCK), TB, 0, stream>>>(
        tmp, nullptr, enc_g, enc_bb, x, N, 1);

    // ---- relative position MLP per edge
    build_rel_kernel<<<cdiv(E, TB), TB, 0, stream>>>(xyz, ei, rel, E);
    launch_gemm<0>(rel, posW1p, pos_b1, eA, E, 32, 128, 128, stream);
    ln128_kernel<<<cdiv(E, WAVES_PER_BLOCK), TB, 0, stream>>>(
        eA, nullptr, pos_g, pos_bb, eA, E, 1);
    launch_gemm<0>(eA, pos_W2, pos_b2, eF, E, 128, 128, 128, stream);

    // ---- transformer layers
    for (int l = 0; l < 3; ++l) {
        const size_t WO = (size_t)l * 128 * 128;
        launch_gemm<0>(x, Wq + WO, bq + l * 128, qb, N, 128, 128, 128, stream);
        launch_gemm<0>(x, Wk + WO, bk + l * 128, kb, N, 128, 128, 128, stream);
        launch_gemm<0>(x, Wv + WO, bv + l * 128, vb, N, 128, 128, 128, stream);
        launch_gemm<0>(eF, We + WO, nullptr, eA, E, 128, 128, 128, stream);

        attn_alpha_kernel<<<cdiv(E * 4, TB), TB, 0, stream>>>(qb, kb, eA, ei, alpha, E);
        init_softmax_kernel<<<cdiv(N * 4, TB), TB, 0, stream>>>(amax, den, N * 4);
        seg_max_kernel<<<cdiv(E * 4, TB), TB, 0, stream>>>(alpha, ei, amax, E);
        seg_expsum_kernel<<<cdiv(E * 4, TB), TB, 0, stream>>>(alpha, ei, amax, den, E);

        zero_kernel<<<(int)(((size_t)N * 128 + TB - 1) / TB), TB, 0, stream>>>(
            agg, (size_t)N * 128);
        attn_msg_kernel<<<(int)(((size_t)E * 128 + TB - 1) / TB), TB, 0, stream>>>(
            vb, eA, alpha, den, ei, agg, E);

        // beta-gated skip
        launch_gemm<0>(x, Wskip + WO, bskip + l * 128, rb, N, 128, 128, 128, stream);
        beta_gate_kernel<<<cdiv(N, WAVES_PER_BLOCK), TB, 0, stream>>>(
            agg, rb, Wbeta + (size_t)l * 384, N);

        // x = LN(gated + x)
        ln128_kernel<<<cdiv(N, WAVES_PER_BLOCK), TB, 0, stream>>>(
            agg, x, ln1_g + l * 128, ln1_b + l * 128, x, N, 0);

        // FFN: gelu(x@fW1+fb1)@fW2+fb2, then x = LN(h + x)
        launch_gemm<2>(x, fW1 + (size_t)l * 128 * 256, fb1 + l * 256, tmp,
                       N, 128, 256, 256, stream);
        launch_gemm<0>(tmp, fW2 + (size_t)l * 256 * 128, fb2 + l * 128, agg,
                       N, 256, 128, 128, stream);
        ln128_kernel<<<cdiv(N, WAVES_PER_BLOCK), TB, 0, stream>>>(
            agg, x, ln2_g + l * 128, ln2_b + l * 128, x, N, 0);
    }

    // ---- heads
    launch_gemm<1>(x, cls_W1, cls_b1, h64, N, 128, 64, 64, stream);
    launch_gemm<0>(h64, clsW2p, cls_b2, cls_out, N, 64, 32, 4, stream);
    launch_gemm<1>(x, box_W1, box_b1, h64, N, 128, 64, 64, stream);
    launch_gemm<0>(h64, boxW2p, box_b2, box_out, N, 64, 32, 7, stream);
}
